// LSTMDecoder_33243046871674
// MI455X (gfx1250) — compile-verified, matrix operand order checked by reading, not run on hardware
//
#include <hip/hip_runtime.h>

// ---------------------------------------------------------------------------
// LSTM decoder for MI455X (gfx1250, wave32, WMMA bf16 16x16x32, f32 accum)
// GEMMs use async global->LDS (ASYNCcnt) double buffering + v_wmma.
// ---------------------------------------------------------------------------

typedef unsigned short u16;
typedef __attribute__((ext_vector_type(16))) __bf16 v16bf;
typedef __attribute__((ext_vector_type(8)))  float  v8f;
typedef __attribute__((ext_vector_type(8)))  unsigned int v8u;

// Problem constants (match reference)
constexpr int Bsz = 512;
constexpr int Isz = 2560;
constexpr int Hsz = 512;
constexpr int Esz = 300;
constexpr int Vsz = 8000;
constexpr int Tsz = 16;
constexpr int Gsz = 4 * Hsz;        // 2048 gates
constexpr int Kx  = Esz + Hsz;      // 812
constexpr int Kxp = 832;            // padded to multiple of 32
constexpr int BT  = Bsz * Tsz;      // 8192
constexpr int Vp  = 8064;           // V padded to multiple of 128
constexpr int SOS = 2;

constexpr int TILE_HALVES = 128 * 32;            // one LDS tile (A or B)

// ---------------------------------------------------------------------------
// helpers
// ---------------------------------------------------------------------------
__device__ __forceinline__ u16 f2bf(float f) {
  unsigned int u = __builtin_bit_cast(unsigned int, f);
  u += 0x7FFFu + ((u >> 16) & 1u);      // round-to-nearest-even
  return (u16)(u >> 16);
}

__device__ __forceinline__ float sigf(float x) {
  return 1.0f / (1.0f + __expf(-x));
}

// Async copy of 16 bytes global -> LDS (tracked by ASYNCcnt).
__device__ __forceinline__ void async_b128(unsigned ldsByteAddr, const u16* gsrc) {
  unsigned long long ga = (unsigned long long)gsrc;
  asm volatile("global_load_async_to_lds_b128 %0, %1, off"
               :: "v"(ldsByteAddr), "v"(ga) : "memory");
}

// Load one 16x32 bf16 fragment from an LDS tile (row stride = 32 halves).
// CDNA5 layout: lane<16 -> row=lane, K 0..7 (v0-3) and 16..23 (v4-7);
//               lane>=16 -> row=lane-16, K 8..15 and 24..31.
__device__ __forceinline__ v16bf frag_lds(const u16* base, int lane) {
  const int r   = lane & 15;
  const int kLo = (lane >> 4) << 3;            // 0 or 8
  const u16* row = base + r * 32;
  uint4 lo = *(const uint4*)(row + kLo);
  uint4 hi = *(const uint4*)(row + kLo + 16);
  v8u t;
  t[0] = lo.x; t[1] = lo.y; t[2] = lo.z; t[3] = lo.w;
  t[4] = hi.x; t[5] = hi.y; t[6] = hi.z; t[7] = hi.w;
  return __builtin_bit_cast(v16bf, t);
}

// ---------------------------------------------------------------------------
// Generic bf16 TN GEMM:  C[M,N] = A[M,K] * Bw[N,K]^T (+ bias[N]) (+ Cadd)
// 128x128x32 tiles, 8 waves (4x2), per-wave 32x64 (2x4 WMMA frags).
// Double-buffered LDS filled with global_load_async_to_lds_b128.
// ---------------------------------------------------------------------------
__global__ __launch_bounds__(256)
void gemm_bf16_wmma(const u16* __restrict__ A,  int lda,
                    const u16* __restrict__ Bw, int ldb,
                    float* __restrict__ C, int ldc,
                    const float* __restrict__ bias,
                    const float* __restrict__ Cadd, int ldadd,
                    int K, int Nreal)
{
  __shared__ u16 lsA[2][TILE_HALVES];
  __shared__ u16 lsB[2][TILE_HALVES];

  const int tid   = threadIdx.x;
  const int lane  = tid & 31;
  const int wid   = tid >> 5;
  const int waveM = wid >> 1;          // 0..3 -> 32 rows each
  const int waveN = wid & 1;           // 0..1 -> 64 cols each
  const long rowBase = (long)blockIdx.y * 128;
  const long colBase = (long)blockIdx.x * 128;

  // LDS byte addresses (flat LDS aperture: addr[31:0] is the LDS offset)
  const unsigned lsA0 = (unsigned)(unsigned long long)&lsA[0][0];
  const unsigned lsB0 = (unsigned)(unsigned long long)&lsB[0][0];

  // This thread's two 16-byte chunks per tile: i = tid, tid+256
  const int r0 = tid >> 2;             // rows 0..63
  const int q0 = tid & 3;
  const int r1 = (tid + 256) >> 2;     // rows 64..127
  const int q1 = q0;

  const u16* gA0 = A  + (rowBase + r0) * (long)lda + q0 * 8;
  const u16* gA1 = A  + (rowBase + r1) * (long)lda + q1 * 8;
  const u16* gB0 = Bw + (colBase + r0) * (long)ldb + q0 * 8;
  const u16* gB1 = Bw + (colBase + r1) * (long)ldb + q1 * 8;

  const unsigned oA0 = (unsigned)((r0 * 32 + q0 * 8) * 2);
  const unsigned oA1 = (unsigned)((r1 * 32 + q1 * 8) * 2);
  const unsigned oB0 = oA0;
  const unsigned oB1 = oA1;

  v8f zero = {};
  v8f acc[2][4];
  #pragma unroll
  for (int mf = 0; mf < 2; ++mf)
    #pragma unroll
    for (int nf = 0; nf < 4; ++nf) acc[mf][nf] = zero;

  const int nK = K >> 5;               // K / 32 slabs

  // Prologue: fill buffer 0 with slab 0
  {
    const unsigned bufOff = 0;
    async_b128(lsA0 + bufOff + oA0, gA0);
    async_b128(lsA0 + bufOff + oA1, gA1);
    async_b128(lsB0 + bufOff + oB0, gB0);
    async_b128(lsB0 + bufOff + oB1, gB1);
  }

  for (int ks = 0; ks < nK; ++ks) {
    const int cur = ks & 1;
    const int nxt = cur ^ 1;

    // All waves finished computing from buf[nxt] (previous iteration)
    __syncthreads();

    if (ks + 1 < nK) {
      const long kOff = (long)(ks + 1) << 5;
      const unsigned bufOff = (unsigned)(nxt * TILE_HALVES * 2);
      async_b128(lsA0 + bufOff + oA0, gA0 + kOff);
      async_b128(lsA0 + bufOff + oA1, gA1 + kOff);
      async_b128(lsB0 + bufOff + oB0, gB0 + kOff);
      async_b128(lsB0 + bufOff + oB1, gB1 + kOff);
      asm volatile("s_wait_asynccnt 0x4" ::: "memory");   // buf[cur] landed
    } else {
      asm volatile("s_wait_asynccnt 0x0" ::: "memory");
    }
    __syncthreads();                    // everyone's buf[cur] data visible

    v16bf af[2], bfr[4];
    #pragma unroll
    for (int mf = 0; mf < 2; ++mf)
      af[mf] = frag_lds(&lsA[cur][0] + (waveM * 32 + mf * 16) * 32, lane);
    #pragma unroll
    for (int nf = 0; nf < 4; ++nf)
      bfr[nf] = frag_lds(&lsB[cur][0] + (waveN * 64 + nf * 16) * 32, lane);

    #pragma unroll
    for (int mf = 0; mf < 2; ++mf)
      #pragma unroll
      for (int nf = 0; nf < 4; ++nf)
        acc[mf][nf] = __builtin_amdgcn_wmma_f32_16x16x32_bf16(
            false, af[mf], false, bfr[nf], (short)0, acc[mf][nf], false, false);
  }

  // Epilogue. D layout: VGPR r -> M = r (+8 for lanes>=16), N = lane&15.
  const int n    = lane & 15;
  const int mOff = (lane >> 4) << 3;   // 0 or 8
  #pragma unroll
  for (int mf = 0; mf < 2; ++mf) {
    #pragma unroll
    for (int nf = 0; nf < 4; ++nf) {
      long col = colBase + waveN * 64 + nf * 16 + n;
      if (col < Nreal) {
        float bv = bias ? bias[col] : 0.0f;
        #pragma unroll
        for (int r = 0; r < 8; ++r) {
          long row = rowBase + waveM * 32 + mf * 16 + mOff + r;
          float v = acc[mf][nf][r] + bv;
          if (Cadd) v += Cadd[row * (long)ldadd + col];
          C[row * (long)ldc + col] = v;
        }
      }
    }
  }
}

// ---------------------------------------------------------------------------
// Small elementwise kernels
// ---------------------------------------------------------------------------
__global__ void fill0_bf(u16* dst, long n) {
  long i = blockIdx.x * (long)blockDim.x + threadIdx.x;
  if (i < n) dst[i] = 0;
}

__global__ void cvt2d_bf(const float* __restrict__ src, u16* __restrict__ dst,
                         long rows, long cols, long lds_, long ldd) {
  long i = blockIdx.x * (long)blockDim.x + threadIdx.x;
  if (i >= rows * cols) return;
  long r = i / cols, c = i - r * cols;
  dst[r * ldd + c] = f2bf(src[r * lds_ + c]);
}

__global__ void addv(const float* a, const float* b, float* o, int n) {
  int i = blockIdx.x * blockDim.x + threadIdx.x;
  if (i < n) o[i] = a[i] + b[i];
}

// x[b,t,k] : k<E -> embedding[word], k<E+H -> context, else pad 0
__global__ void build_x(const int* __restrict__ targets,
                        const float* __restrict__ emb,
                        const float* __restrict__ ctx32,
                        u16* __restrict__ x) {
  long i = blockIdx.x * (long)blockDim.x + threadIdx.x;
  if (i >= (long)BT * Kxp) return;
  int  k  = (int)(i % Kxp);
  long bt = i / Kxp;
  int  t  = (int)(bt % Tsz);
  int  b  = (int)(bt / Tsz);
  float v;
  if (k < Esz) {
    int w = (t == 0) ? SOS : targets[b * Tsz + (t - 1)];
    v = emb[(long)w * Esz + k];
  } else if (k < Kx) {
    v = ctx32[(long)b * Hsz + (k - Esz)];
  } else {
    v = 0.0f;
  }
  x[i] = f2bf(v);
}

// h0=h1=context (bf16 into hcat halves), c0=c1=cell0
__global__ void init_state(const float* __restrict__ ctx32,
                           const float* __restrict__ cell032,
                           u16* __restrict__ hcat,
                           float* __restrict__ c0, float* __restrict__ c1) {
  int i = blockIdx.x * blockDim.x + threadIdx.x;
  if (i >= Bsz * Hsz) return;
  int b = i >> 9, h = i & (Hsz - 1);
  u16 hb = f2bf(ctx32[i]);
  hcat[(long)b * (2 * Hsz) + h]       = hb;
  hcat[(long)b * (2 * Hsz) + Hsz + h] = hb;
  c0[i] = cell032[i];
  c1[i] = cell032[i];
}

// gates [B,4H] order i,f,g,o ; c in/out fp32 ; h -> bf16 (strided) (+hs copy)
__global__ void lstm_cell(const float* __restrict__ gates,
                          float* __restrict__ c,
                          u16* __restrict__ hout, int hstride,
                          u16* __restrict__ hs, int t) {
  int i = blockIdx.x * blockDim.x + threadIdx.x;
  if (i >= Bsz * Hsz) return;
  int b = i >> 9, h = i & (Hsz - 1);
  const float* g = gates + (long)b * Gsz;
  float iG = sigf(g[h]);
  float fG = sigf(g[Hsz + h]);
  float gG = tanhf(g[2 * Hsz + h]);
  float oG = sigf(g[3 * Hsz + h]);
  float cn = fG * c[i] + iG * gG;
  c[i] = cn;
  u16 hb = f2bf(oG * tanhf(cn));
  hout[(long)b * hstride + h] = hb;
  if (hs) hs[((long)b * Tsz + t) * Hsz + h] = hb;
}

// ---------------------------------------------------------------------------
// Host side
// ---------------------------------------------------------------------------
static inline long cdiv(long a, long b) { return (a + b - 1) / b; }

extern "C" void kernel_launch(void* const* d_in, const int* in_sizes, int n_in,
                              void* d_out, int out_size, void* d_ws, size_t ws_size,
                              hipStream_t stream) {
  const float* ff     = (const float*)d_in[0];
  const int*   tgt    = (const int*)  d_in[1];
  const float* emb    = (const float*)d_in[3];
  const float* Win    = (const float*)d_in[4];
  const float* b_in   = (const float*)d_in[5];
  const float* Wcell  = (const float*)d_in[6];
  const float* b_cell = (const float*)d_in[7];
  const float* Wih0   = (const float*)d_in[8];
  const float* Whh0   = (const float*)d_in[9];
  const float* b_ih0  = (const float*)d_in[10];
  const float* b_hh0  = (const float*)d_in[11];
  const float* Wih1   = (const float*)d_in[12];
  const float* Whh1   = (const float*)d_in[13];
  const float* b_ih1  = (const float*)d_in[14];
  const float* b_hh1  = (const float*)d_in[15];
  const float* Wout   = (const float*)d_in[16];
  const float* b_out  = (const float*)d_in[17];
  float* logits = (float*)d_out;

  // workspace carve-up (256B aligned)
  char* wp = (char*)d_ws;
  auto alloc = [&](size_t bytes) -> void* {
    void* r = (void*)wp;
    wp += (bytes + 255) & ~(size_t)255;
    return r;
  };
  u16* ff_bf    = (u16*)alloc((size_t)Bsz * Isz * 2);
  u16* Win_bf   = (u16*)alloc((size_t)Hsz * Isz * 2);
  u16* Wcell_bf = (u16*)alloc((size_t)Hsz * Isz * 2);
  u16* Wih0_bf  = (u16*)alloc((size_t)Gsz * Kxp * 2);
  u16* Whh0_bf  = (u16*)alloc((size_t)Gsz * Hsz * 2);
  u16* Wcat1_bf = (u16*)alloc((size_t)Gsz * (2 * Hsz) * 2);
  u16* Wout_bf  = (u16*)alloc((size_t)Vp * Hsz * 2);
  u16* x_bf     = (u16*)alloc((size_t)BT * Kxp * 2);
  u16* hs_bf    = (u16*)alloc((size_t)BT * Hsz * 2);
  u16* hcat     = (u16*)alloc((size_t)Bsz * (2 * Hsz) * 2);
  float* ctx32  = (float*)alloc((size_t)Bsz * Hsz * 4);
  float* cl032  = (float*)alloc((size_t)Bsz * Hsz * 4);
  float* c0     = (float*)alloc((size_t)Bsz * Hsz * 4);
  float* c1     = (float*)alloc((size_t)Bsz * Hsz * 4);
  float* bias0  = (float*)alloc((size_t)Gsz * 4);
  float* bias1  = (float*)alloc((size_t)Gsz * 4);
  float* xg0    = (float*)alloc((size_t)BT * Gsz * 4);
  float* gates  = (float*)alloc((size_t)Bsz * Gsz * 4);

  const int TB = 256;
  auto blocks1d = [&](long n) { return dim3((unsigned)cdiv(n, TB)); };

  // --- bf16 conversions / padding ---
  fill0_bf<<<blocks1d((long)Gsz * Kxp), TB, 0, stream>>>(Wih0_bf, (long)Gsz * Kxp);
  fill0_bf<<<blocks1d((long)Vp * Hsz),  TB, 0, stream>>>(Wout_bf, (long)Vp * Hsz);
  cvt2d_bf<<<blocks1d((long)Bsz * Isz), TB, 0, stream>>>(ff,    ff_bf,    Bsz, Isz, Isz, Isz);
  cvt2d_bf<<<blocks1d((long)Hsz * Isz), TB, 0, stream>>>(Win,   Win_bf,   Hsz, Isz, Isz, Isz);
  cvt2d_bf<<<blocks1d((long)Hsz * Isz), TB, 0, stream>>>(Wcell, Wcell_bf, Hsz, Isz, Isz, Isz);
  cvt2d_bf<<<blocks1d((long)Gsz * Kx),  TB, 0, stream>>>(Wih0,  Wih0_bf,  Gsz, Kx,  Kx,  Kxp);
  cvt2d_bf<<<blocks1d((long)Gsz * Hsz), TB, 0, stream>>>(Whh0,  Whh0_bf,  Gsz, Hsz, Hsz, Hsz);
  cvt2d_bf<<<blocks1d((long)Gsz * Hsz), TB, 0, stream>>>(Wih1,  Wcat1_bf,       Gsz, Hsz, Hsz, 2 * Hsz);
  cvt2d_bf<<<blocks1d((long)Gsz * Hsz), TB, 0, stream>>>(Whh1,  Wcat1_bf + Hsz, Gsz, Hsz, Hsz, 2 * Hsz);
  cvt2d_bf<<<blocks1d((long)Vsz * Hsz), TB, 0, stream>>>(Wout,  Wout_bf,  Vsz, Hsz, Hsz, Hsz);
  addv<<<blocks1d(Gsz), TB, 0, stream>>>(b_ih0, b_hh0, bias0, Gsz);
  addv<<<blocks1d(Gsz), TB, 0, stream>>>(b_ih1, b_hh1, bias1, Gsz);

  // --- context / cell0: [B,H] = ff @ W^T + b ---
  {
    dim3 g(Hsz / 128, Bsz / 128);
    gemm_bf16_wmma<<<g, TB, 0, stream>>>(ff_bf, Isz, Win_bf,   Isz, ctx32, Hsz,
                                         b_in,   nullptr, 0, Isz, Hsz);
    gemm_bf16_wmma<<<g, TB, 0, stream>>>(ff_bf, Isz, Wcell_bf, Isz, cl032, Hsz,
                                         b_cell, nullptr, 0, Isz, Hsz);
  }
  init_state<<<blocks1d((long)Bsz * Hsz), TB, 0, stream>>>(ctx32, cl032, hcat, c0, c1);

  // --- x = [emb(words) | context] (bf16, K padded) ---
  build_x<<<blocks1d((long)BT * Kxp), TB, 0, stream>>>(tgt, emb, ctx32, x_bf);

  // --- xg0[b*T+t, 4H] = x @ W_ih0^T + (b_ih0 + b_hh0) ---
  {
    dim3 g(Gsz / 128, BT / 128);
    gemm_bf16_wmma<<<g, TB, 0, stream>>>(x_bf, Kxp, Wih0_bf, Kxp, xg0, Gsz,
                                         bias0, nullptr, 0, Kxp, Gsz);
  }

  // --- recurrence over T ---
  for (int t = 0; t < Tsz; ++t) {
    dim3 g(Gsz / 128, Bsz / 128);
    // layer 0: gates = xg0[:,t,:] + h0 @ W_hh0^T
    gemm_bf16_wmma<<<g, TB, 0, stream>>>(hcat, 2 * Hsz, Whh0_bf, Hsz, gates, Gsz,
                                         nullptr, xg0 + (long)t * Gsz, Tsz * Gsz,
                                         Hsz, Gsz);
    lstm_cell<<<blocks1d((long)Bsz * Hsz), TB, 0, stream>>>(
        gates, c0, hcat, 2 * Hsz, nullptr, t);
    // layer 1: gates = [h0,h1] @ [W_ih1|W_hh1]^T + (b_ih1 + b_hh1)
    gemm_bf16_wmma<<<g, TB, 0, stream>>>(hcat, 2 * Hsz, Wcat1_bf, 2 * Hsz, gates, Gsz,
                                         bias1, nullptr, 0, 2 * Hsz, Gsz);
    lstm_cell<<<blocks1d((long)Bsz * Hsz), TB, 0, stream>>>(
        gates, c1, hcat + Hsz, 2 * Hsz, hs_bf, t);
  }

  // --- logits[b*T+t, V] = hs @ W_out^T + b_out ---
  {
    dim3 g(Vp / 128, BT / 128);   // 63 x 64 tiles, col-guarded at V=8000
    gemm_bf16_wmma<<<g, TB, 0, stream>>>(hs_bf, Hsz, Wout_bf, Hsz, logits, Vsz,
                                         b_out, nullptr, 0, Hsz, Vsz);
  }
}
